// MultitaskHeads_23493471109247
// MI455X (gfx1250) — matrix operand order repaired and checked by reading, not compiled, not run on hardware
//
#include <hip/hip_runtime.h>
#include <hip/hip_bf16.h>

// ---------------- problem constants ----------------
#define BB 8
#define SS 512
#define NTOK (BB * SS)      // 4096 tokens
#define DD 1024             // d_model
#define FF 2048             // head hidden
#define HH 16               // heads
#define AT_STRIDE (DD + 8)  // LDS A-tile row stride (ushorts): +16B pad -> bank spread
// ---------------------------------------------------

typedef __bf16        v16bf __attribute__((ext_vector_type(16)));
typedef float         v8f   __attribute__((ext_vector_type(8)));
typedef unsigned int  u32x4 __attribute__((ext_vector_type(4)));

union Frag {            // 32 bytes: WMMA A/B operand = 16 bf16 = two b128 loads
    v16bf v;
    u32x4 q[2];
};

// CDNA5 async global->LDS DMA (GV mode), tracked with ASYNCcnt.
#define ASYNC_LOAD_B128(ldsoff, gaddr)                                  \
    asm volatile("global_load_async_to_lds_b128 %0, %1, off"            \
                 :: "v"(ldsoff), "v"(gaddr) : "memory")
#define ASYNC_WAIT() asm volatile("s_wait_asynccnt 0x0" ::: "memory")

// Hard scheduling fence: machine order == source order across this point.
#define SCHED_FENCE() __builtin_amdgcn_sched_barrier(0)

static __device__ __forceinline__ unsigned short f32_to_bf16(float f) {
    unsigned u = __float_as_uint(f);
    u += 0x7FFFu + ((u >> 16) & 1u);     // round-to-nearest-even
    return (unsigned short)(u >> 16);
}

static __device__ __forceinline__ float gelu_tanh(float x) {
    const float k = 0.7978845608028654f;
    float x3 = x * x * x;
    return 0.5f * x * (1.0f + tanhf(k * (x + 0.044715f * x3)));
}

// ============================================================
// Kernel 1: masked embedding sum -> X bf16  [NTOK, DD]
// ============================================================
__global__ void build_x_kernel(const int* __restrict__ selfies,
                               const int* __restrict__ tasks,
                               const float* __restrict__ values,
                               const int* __restrict__ pmask,
                               const float* __restrict__ emb_s,
                               const float* __restrict__ emb_t,
                               const float* __restrict__ val_vec,
                               unsigned short* __restrict__ X) {
    const int tok = blockIdx.x;
    const int sid = selfies[tok];
    const int tid = tasks[tok];
    const float val = values[tok];
    const int live = pmask[tok];
    const float* es = emb_s + (size_t)sid * DD;
    const float* et = emb_t + (size_t)tid * DD;
    unsigned short* xr = X + (size_t)tok * DD;
#pragma unroll
    for (int j = 0; j < 4; ++j) {
        int d = threadIdx.x + j * 256;
        float h = live ? (es[d] + et[d] + val * val_vec[d]) : 0.0f;
        xr[d] = f32_to_bf16(h);
    }
}

// ============================================================
// Kernel 2: tiled transpose + fp32->bf16. out[N][K] = in[K][N]
// (batched over gridDim.z for W1)
// ============================================================
__global__ void transpose_bf16_kernel(const float* __restrict__ in,
                                      unsigned short* __restrict__ out,
                                      int K, int N) {
    __shared__ float tile[32][33];
    const size_t base = (size_t)blockIdx.z * (size_t)K * (size_t)N;
    int k = blockIdx.y * 32 + threadIdx.y;
    int n = blockIdx.x * 32 + threadIdx.x;
    tile[threadIdx.y][threadIdx.x] = in[base + (size_t)k * N + n];
    __syncthreads();
    int no = blockIdx.x * 32 + threadIdx.y;
    int ko = blockIdx.y * 32 + threadIdx.x;
    out[base + (size_t)no * K + ko] = f32_to_bf16(tile[threadIdx.x][threadIdx.y]);
}

// ---- operand fetch / compute macros shared by both GEMM kernels ----
// (expand against local names: Alds, Brow[4], half, c0..c3)
#define LOAD_AB(aF, bF, kk)                                                   \
    do {                                                                      \
        aF.q[0] = *reinterpret_cast<const u32x4*>(Alds + (kk) + half * 8);    \
        aF.q[1] = *reinterpret_cast<const u32x4*>(Alds + (kk) + 16 + half * 8); \
        _Pragma("unroll")                                                     \
        for (int j_ = 0; j_ < 4; ++j_) {                                      \
            bF[j_].q[0] = *reinterpret_cast<const u32x4*>(Brow[j_] + (kk) + half * 8);      \
            bF[j_].q[1] = *reinterpret_cast<const u32x4*>(Brow[j_] + (kk) + 16 + half * 8); \
        }                                                                     \
    } while (0)

#define WMMA4(aF, bF)                                                                              \
    do {                                                                                           \
        c0 = __builtin_amdgcn_wmma_f32_16x16x32_bf16(false, aF.v, false, bF[0].v, (short)0, c0, false, false); \
        c1 = __builtin_amdgcn_wmma_f32_16x16x32_bf16(false, aF.v, false, bF[1].v, (short)0, c1, false, false); \
        c2 = __builtin_amdgcn_wmma_f32_16x16x32_bf16(false, aF.v, false, bF[2].v, (short)0, c2, false, false); \
        c3 = __builtin_amdgcn_wmma_f32_16x16x32_bf16(false, aF.v, false, bF[3].v, (short)0, c3, false, false); \
    } while (0)

// Fenced ping-pong pipeline: loads of step k+32 are pinned BEFORE the
// WMMAs of step k, so each WMMA group has a full half-iteration (10
// independent b128/ds loads) of latency cover, and RA must keep two
// operand sets live (staggered s_wait instead of wait-to-zero).
#define KLOOP_PIPELINED()                                                     \
    do {                                                                      \
        LOAD_AB(aA, bA, 0);                                                   \
        SCHED_FENCE();                                                        \
        for (int k = 0; k < DD; k += 64) {                                    \
            LOAD_AB(aB, bB, k + 32);                                          \
            SCHED_FENCE();                                                    \
            WMMA4(aA, bA);                                                    \
            SCHED_FENCE();                                                    \
            LOAD_AB(aA, bA, (k + 64 < DD) ? (k + 64) : 0);                    \
            SCHED_FENCE();                                                    \
            WMMA4(aB, bB);                                                    \
            SCHED_FENCE();                                                    \
        }                                                                     \
    } while (0)

// ============================================================
// Kernel 3: shared = gelu(X @ Ws + bs)  in bf16.
// Block = 4 waves; 16-row A tile async-DMA'd into LDS once, shared by
// all waves. Each wave owns FOUR 16x16 n-tiles. WsT is bf16 [N][K].
// ============================================================
__global__ void shared_gemm_kernel(const unsigned short* __restrict__ X,
                                   const unsigned short* __restrict__ WsT,
                                   const float* __restrict__ bs,
                                   unsigned short* __restrict__ Sh) {
    __shared__ unsigned short Atile[16 * AT_STRIDE];   // ~33 KB

    const int rowtile = blockIdx.x;                    // 256 tiles of 16 rows

    // ---- async stage of the 16x1024 bf16 A tile (ASYNCcnt path) ----
    {
        const int r = threadIdx.x >> 3;                // 16 rows, 8 threads/row
        const int c = threadIdx.x & 7;
        const unsigned short* src = X + (size_t)(rowtile * 16 + r) * DD;
        const unsigned ldsrow = (unsigned)(uintptr_t)(Atile + r * AT_STRIDE);
#pragma unroll
        for (int j = 0; j < 16; ++j) {                 // 128 units of 16 B per row
            int unit = c + 8 * j;
            ASYNC_LOAD_B128(ldsrow + unit * 16,
                            (unsigned long long)(uintptr_t)(src + unit * 8));
        }
    }
    ASYNC_WAIT();
    __syncthreads();

    const int wave = threadIdx.x >> 5;
    const int lane = threadIdx.x & 31;
    const int half = lane >> 4;
    const int col  = lane & 15;
    const int nb   = (blockIdx.y * 4 + wave) * 4;      // first of 4 n-tiles

    const unsigned short* Alds = Atile + (size_t)col * AT_STRIDE;
    const unsigned short* Brow[4];
#pragma unroll
    for (int j = 0; j < 4; ++j)
        Brow[j] = WsT + (size_t)((nb + j) * 16 + col) * DD;

    v8f c0 = {}, c1 = {}, c2 = {}, c3 = {};
    Frag aA, aB;
    Frag bA[4], bB[4];
    KLOOP_PIPELINED();

    v8f acc[4] = {c0, c1, c2, c3};
#pragma unroll
    for (int j = 0; j < 4; ++j) {
        const int n = (nb + j) * 16 + col;
        const float bias = bs[n];
#pragma unroll
        for (int r = 0; r < 8; ++r) {
            int m = rowtile * 16 + r + half * 8;       // C/D layout: row = r + 8*half
            Sh[(size_t)m * DD + n] = f32_to_bf16(gelu_tanh(acc[j][r] + bias));
        }
    }
}

// ============================================================
// Kernel 4a/4b: routing. counts[h]=0 ; then per-token scatter.
// ============================================================
__global__ void zero_counts_kernel(int* counts) {
    if (threadIdx.x < HH) counts[threadIdx.x] = 0;
}

__global__ void route_kernel(const int* __restrict__ tasks,
                             int* __restrict__ counts,
                             int* __restrict__ tok_list,
                             float* __restrict__ out) {
    const int t = blockIdx.x * 256 + threadIdx.x;
    const int task = tasks[t];
    if (task > 0) {
        int h = (task - 1) & (HH - 1);
        int slot = atomicAdd(&counts[h], 1);
        tok_list[h * NTOK + slot] = t;
    } else {
        out[t] = 0.0f;                // invalid token -> 0 (O == 1)
    }
}

// ============================================================
// Kernel 5: per-head gathered MLP.
//   hid = gelu(shared[tok] @ W1[h] + b1[h]) ; out = hid . W2[h] + b2[h]
// Block = (head h, 16-token tile): gathered A tile async-DMA'd into LDS
// once, shared by all 8 waves. Each wave covers 16 n-tiles in 4 groups
// of 4 accumulators. Second GEMM (F->O=1) fused as per-lane FMA +
// ds_add_f32 reduction. W1T is bf16 [H][F][D].
// ============================================================
__global__ void head_mlp_kernel(const unsigned short* __restrict__ Sh,
                                const unsigned short* __restrict__ W1T,
                                const float* __restrict__ b1,
                                const float* __restrict__ W2,
                                const float* __restrict__ b2,
                                const int* __restrict__ counts,
                                const int* __restrict__ tok_list,
                                float* __restrict__ out) {
    __shared__ unsigned short Atile[16 * AT_STRIDE];   // ~33 KB
    __shared__ float partial[16];
    if (threadIdx.x < 16) partial[threadIdx.x] = 0.0f;

    const int h    = blockIdx.x >> 8;
    const int tile = blockIdx.x & 255;
    const int cnt  = counts[h];
    __syncthreads();
    if (tile * 16 >= cnt) return;                      // block-uniform exit

    // ---- async gather of 16 token rows (invalid rows -> 0) ----
    {
        const int r = threadIdx.x >> 4;                // 16 rows, 16 threads/row
        const int c = threadIdx.x & 15;
        const int ti = tile * 16 + r;
        const int tok = (ti < cnt) ? tok_list[h * NTOK + ti] : -1;
        const unsigned ldsrow = (unsigned)(uintptr_t)(Atile + r * AT_STRIDE);
        if (tok >= 0) {
            const unsigned short* src = Sh + (size_t)tok * DD;
#pragma unroll
            for (int j = 0; j < 8; ++j) {              // 8 units of 16 B per thread
                int unit = c + 16 * j;
                ASYNC_LOAD_B128(ldsrow + unit * 16,
                                (unsigned long long)(uintptr_t)(src + unit * 8));
            }
        } else {
            const u32x4 zq = {0u, 0u, 0u, 0u};
            u32x4* dst = reinterpret_cast<u32x4*>(Atile + r * AT_STRIDE);
#pragma unroll
            for (int j = 0; j < 8; ++j)
                dst[c + 16 * j] = zq;
        }
    }
    ASYNC_WAIT();
    __syncthreads();

    const int wave = threadIdx.x >> 5;
    const int lane = threadIdx.x & 31;
    const int half = lane >> 4;
    const int col  = lane & 15;

    const unsigned short* Alds = Atile + (size_t)col * AT_STRIDE;
    const unsigned short* Wh = W1T + (size_t)h * FF * DD;

    float rowacc[8];
#pragma unroll
    for (int r = 0; r < 8; ++r) rowacc[r] = 0.0f;

    for (int g = 0; g < 4; ++g) {                      // 4 groups of 4 n-tiles
        const int ntb = wave * 16 + g * 4;             // wave covers tiles [w*16, w*16+16)
        const unsigned short* Brow[4];
#pragma unroll
        for (int j = 0; j < 4; ++j)
            Brow[j] = Wh + (size_t)((ntb + j) * 16 + col) * DD;
        __builtin_prefetch(Brow[3] + 16 * DD, 0, 0);   // next group's panel

        v8f c0 = {}, c1 = {}, c2 = {}, c3 = {};
        Frag aA, aB;
        Frag bA[4], bB[4];
        KLOOP_PIPELINED();

        v8f acc[4] = {c0, c1, c2, c3};
#pragma unroll
        for (int j = 0; j < 4; ++j) {
            const int n = (ntb + j) * 16 + col;
            const float b1v = b1[h * FF + n];
            const float w2v = W2[h * FF + n];          // O == 1
#pragma unroll
            for (int r = 0; r < 8; ++r)
                rowacc[r] += gelu_tanh(acc[j][r] + b1v) * w2v;
        }
    }

#pragma unroll
    for (int r = 0; r < 8; ++r)
        atomicAdd(&partial[r + half * 8], rowacc[r]);  // ds_add_f32
    __syncthreads();

    if (threadIdx.x < 16) {
        int ti = tile * 16 + threadIdx.x;
        if (ti < cnt) {
            int tok = tok_list[h * NTOK + ti];
            out[tok] = partial[threadIdx.x] + b2[h];
        }
    }
}

// ============================================================
// Host-side launcher
// ============================================================
extern "C" void kernel_launch(void* const* d_in, const int* in_sizes, int n_in,
                              void* d_out, int out_size, void* d_ws, size_t ws_size,
                              hipStream_t stream) {
    const int*   selfies  = (const int*)  d_in[0];
    const int*   tasks    = (const int*)  d_in[1];
    const float* values   = (const float*)d_in[2];
    const int*   pmask    = (const int*)  d_in[3];
    const float* emb_s    = (const float*)d_in[4];
    const float* emb_t    = (const float*)d_in[5];
    const float* val_vec  = (const float*)d_in[6];
    const float* Ws       = (const float*)d_in[7];
    const float* bs       = (const float*)d_in[8];
    const float* W1       = (const float*)d_in[9];
    const float* b1       = (const float*)d_in[10];
    const float* W2       = (const float*)d_in[11];
    const float* b2       = (const float*)d_in[12];
    float* out = (float*)d_out;

    // ---- workspace layout (bytes) ----
    char* ws = (char*)d_ws;
    const size_t SZ_X   = (size_t)NTOK * DD * 2;        //  8 MB bf16 features
    const size_t SZ_SH  = (size_t)NTOK * DD * 2;        //  8 MB bf16 shared
    const size_t SZ_WST = (size_t)DD * DD * 2;          //  2 MB bf16 Ws^T
    const size_t SZ_W1T = (size_t)HH * FF * DD * 2;     // 64 MB bf16 W1^T
    unsigned short* X    = (unsigned short*)(ws);
    unsigned short* Sh   = (unsigned short*)(ws + SZ_X);
    unsigned short* WsT  = (unsigned short*)(ws + SZ_X + SZ_SH);
    unsigned short* W1T  = (unsigned short*)(ws + SZ_X + SZ_SH + SZ_WST);
    int* counts          = (int*)(ws + SZ_X + SZ_SH + SZ_WST + SZ_W1T);
    int* tok_list        = counts + 64;                 // padded past counts

    // 1) features
    build_x_kernel<<<NTOK, 256, 0, stream>>>(selfies, tasks, values, pmask,
                                             emb_s, emb_t, val_vec, X);
    // 2) weight transposes (fp32 -> bf16, [K][N] -> [N][K])
    {
        dim3 g(DD / 32, DD / 32, 1), b(32, 32, 1);
        transpose_bf16_kernel<<<g, b, 0, stream>>>(Ws, WsT, DD, DD);
    }
    {
        dim3 g(FF / 32, DD / 32, HH), b(32, 32, 1);
        transpose_bf16_kernel<<<g, b, 0, stream>>>(W1, W1T, DD, FF);
    }
    // 3) shared = gelu(X @ Ws + bs): 4 waves x 4 n-tiles = 256 cols/block
    {
        dim3 g(NTOK / 16, DD / 256, 1);
        shared_gemm_kernel<<<g, 128, 0, stream>>>(X, WsT, bs, Sh);
    }
    // 4) routing
    zero_counts_kernel<<<1, 32, 0, stream>>>(counts);
    route_kernel<<<NTOK / 256, 256, 0, stream>>>(tasks, counts, tok_list, out);
    // 5) per-head gathered MLP + fused projection
    head_mlp_kernel<<<HH * 256, 256, 0, stream>>>(Sh, W1T, b1, W2, b2,
                                                  counts, tok_list, out);
}
